// L2TransformerSequencePool_37830071943285
// MI455X (gfx1250) — matrix operand order repaired and checked by reading, not compiled, not run on hardware
//
#include <hip/hip_runtime.h>
#include <hip/hip_bf16.h>
#include <math.h>

typedef __attribute__((ext_vector_type(16))) _Float16 v16h;
typedef __attribute__((ext_vector_type(8)))  _Float16 v8h;
typedef __attribute__((ext_vector_type(4)))  _Float16 v4h;
typedef __attribute__((ext_vector_type(8)))  float    v8f;

#define TB 256
#define BM 128
#define BN 128
#define BK 32
#define LDW 40   // padded LDS row stride in halves (80B rows -> aligned chunks)

#define GF_RELU  1
#define GF_ACC   2
#define GF_TRANS 4

// ---------------------------------------------------------------------------
// Generic GEMM: C[M,N] (or C[N,M] if GF_TRANS) = act(alpha*A[M,K]@B[N,K]^T + bias[+C])
// f16 WMMA (v_wmma_f32_16x16x32_f16), f32 accumulate. 8 wave32 waves per block,
// 128x128x32 tile, each wave owns a 32x64 sub-tile (2x4 WMMA accumulators).
// Double-buffered LDS, one barrier per K-step: global loads for tile t+1 are
// issued first, the 8 WMMAs run out of buffer `cur` while the fetched tile is
// stored into `cur^1`. Tile staging is fully branch-free: clamped addresses,
// and masking is done with float-side selects (v_cndmask_b32) before the
// f32->f16 convert, so no EXEC divergence wraps vmem/ds ops or the WMMAs.
// ---------------------------------------------------------------------------
__global__ __launch_bounds__(TB)
void gemm_f16_wmma(const float* __restrict__ A, long long lda, long long sA,
                   const float* __restrict__ B, long long ldb, long long sB,
                   float* __restrict__ C, long long ldc, long long sC,
                   const float* __restrict__ bias,
                   int M, int N, int K, float alpha, int flags)
{
  __shared__ _Float16 As[2][BM * LDW];
  __shared__ _Float16 Bs[2][BM * LDW];

  const int tid  = threadIdx.x;
  const int lane = tid & 31;
  const int wave = tid >> 5;
  const int wm   = wave >> 1;   // 0..3 : 32-row group
  const int wn   = wave & 1;    // 0..1 : 64-col group
  const int lh   = lane >> 4;   // 0/1
  const int l16  = lane & 15;

  A += (long long)blockIdx.z * sA;
  B += (long long)blockIdx.z * sB;
  C += (long long)blockIdx.z * sC;

  const int m0 = blockIdx.y * BM;
  const int n0 = blockIdx.x * BN;

  // Uniform (scalar-branch) fast path: float4 global loads + packed LDS stores.
  const bool vec = ((lda & 3) == 0) && ((ldb & 3) == 0) && ((K & (BK - 1)) == 0) &&
                   ((((unsigned long long)(const void*)A) & 15) == 0) &&
                   ((((unsigned long long)(const void*)B) & 15) == 0);

  v8f zero;
  #pragma unroll
  for (int i = 0; i < 8; ++i) zero[i] = 0.0f;
  v8f acc[2][4];
  #pragma unroll
  for (int i = 0; i < 2; ++i)
    #pragma unroll
    for (int j = 0; j < 4; ++j) acc[i][j] = zero;

  // fragment gather + 8 WMMAs out of one LDS buffer pair
  auto do_tile = [&](const _Float16* as, const _Float16* bs) {
    // A frag (ISA 7.12.2, 16-bit A 16x32): lane<16 -> K {0..7,16..23}, lane>=16 -> +8
    v16h af[2], bf[4];
    #pragma unroll
    for (int tm = 0; tm < 2; ++tm) {
      int row = wm * 32 + tm * 16 + l16;
      const v8h lo = *reinterpret_cast<const v8h*>(&as[row * LDW + lh * 8]);
      const v8h hi = *reinterpret_cast<const v8h*>(&as[row * LDW + lh * 8 + 16]);
      af[tm] = __builtin_shufflevector(lo, hi, 0,1,2,3,4,5,6,7,8,9,10,11,12,13,14,15);
    }
    // B frag: lane = column n, lanes 0-15 hold K {0..7,16..23}, lanes 16-31 hold K +8
    #pragma unroll
    for (int tn = 0; tn < 4; ++tn) {
      int col = wn * 64 + tn * 16 + l16;
      const v8h lo = *reinterpret_cast<const v8h*>(&bs[col * LDW + lh * 8]);
      const v8h hi = *reinterpret_cast<const v8h*>(&bs[col * LDW + lh * 8 + 16]);
      bf[tn] = __builtin_shufflevector(lo, hi, 0,1,2,3,4,5,6,7,8,9,10,11,12,13,14,15);
    }
    #pragma unroll
    for (int tm = 0; tm < 2; ++tm)
      #pragma unroll
      for (int tn = 0; tn < 4; ++tn)
        acc[tm][tn] = __builtin_amdgcn_wmma_f32_16x16x32_f16(
            false, af[tm], false, bf[tn], (short)0, acc[tm][tn], false, false);
  };

  if (vec) {
    // --- vector path: 1024 float4 per 128x32 tile; f=i*256+tid, r=f>>3, c4=f&7
    const int vc4 = (tid & 7) * 4;
    long long rA4[4], rB4[4];
    float fmA[4], fmB[4];       // 1.0 / 0.0 row masks (float-side select)
    #pragma unroll
    for (int i = 0; i < 4; ++i) {
      int r  = (i * TB + tid) >> 3;
      int gm = m0 + r;
      int gn = n0 + r;
      bool ma = (gm < M), mb = (gn < N);
      fmA[i] = ma ? 1.0f : 0.0f;
      fmB[i] = mb ? 1.0f : 0.0f;
      rA4[i] = (long long)(ma ? gm : (M - 1)) * lda;
      rB4[i] = (long long)(mb ? gn : (N - 1)) * ldb;
    }
    auto loadv = [&](int kt, float4* va, float4* vb) {
      #pragma unroll
      for (int i = 0; i < 4; ++i) {
        va[i] = *reinterpret_cast<const float4*>(&A[rA4[i] + kt + vc4]);
        vb[i] = *reinterpret_cast<const float4*>(&B[rB4[i] + kt + vc4]);
      }
    };
    auto storev = [&](_Float16* as, _Float16* bs, const float4* va, const float4* vb) {
      #pragma unroll
      for (int i = 0; i < 4; ++i) {
        int r = (i * TB + tid) >> 3;
        v4h ha, hb;
        ha[0] = (_Float16)(va[i].x * fmA[i]);
        ha[1] = (_Float16)(va[i].y * fmA[i]);
        ha[2] = (_Float16)(va[i].z * fmA[i]);
        ha[3] = (_Float16)(va[i].w * fmA[i]);
        hb[0] = (_Float16)(vb[i].x * fmB[i]);
        hb[1] = (_Float16)(vb[i].y * fmB[i]);
        hb[2] = (_Float16)(vb[i].z * fmB[i]);
        hb[3] = (_Float16)(vb[i].w * fmB[i]);
        *reinterpret_cast<v4h*>(&as[r * LDW + vc4]) = ha;   // ds_store_b64
        *reinterpret_cast<v4h*>(&bs[r * LDW + vc4]) = hb;
      }
    };

    const int nt = K / BK;
    float4 va[4], vb[4];
    loadv(0, va, vb);
    storev(As[0], Bs[0], va, vb);
    __syncthreads();
    for (int t = 0; t < nt; ++t) {
      const int cur = t & 1;
      const bool more = (t + 1) < nt;
      if (more) loadv((t + 1) * BK, va, vb);
      if (t + 2 < nt) {  // hint tile t+2 toward the caches (global_prefetch_b8)
        __builtin_prefetch(&A[rA4[tid & 3] + (t + 2) * BK], 0, 1);
        __builtin_prefetch(&B[rB4[tid & 3] + (t + 2) * BK], 0, 1);
      }
      do_tile(As[cur], Bs[cur]);
      if (more) {
        storev(As[cur ^ 1], Bs[cur ^ 1], va, vb);
        __syncthreads();
      }
    }
  } else {
    // --- scalar path (odd strides / ragged K): column fixed per thread ---
    const int sc = tid & 31;
    const int rb = tid >> 5;
    long long rAo[16], rBo[16];
    float fmA[16], fmB[16];
    #pragma unroll
    for (int i = 0; i < 16; ++i) {
      int gm = m0 + rb + i * 8;
      int gn = n0 + rb + i * 8;
      bool ma = (gm < M), mb = (gn < N);
      fmA[i] = ma ? 1.0f : 0.0f;
      fmB[i] = mb ? 1.0f : 0.0f;
      rAo[i] = (long long)(ma ? gm : (M - 1)) * lda;
      rBo[i] = (long long)(mb ? gn : (N - 1)) * ldb;
    }
    auto loads = [&](int kt, float* va, float* vb) {
      int gk  = kt + sc;
      int gkc = (gk < K) ? gk : (K - 1);
      #pragma unroll
      for (int i = 0; i < 16; ++i) {
        va[i] = A[rAo[i] + gkc];
        vb[i] = B[rBo[i] + gkc];
      }
    };
    auto stores = [&](_Float16* as, _Float16* bs, int kt, const float* va, const float* vb) {
      float kmask = ((kt + sc) < K) ? 1.0f : 0.0f;
      #pragma unroll
      for (int i = 0; i < 16; ++i) {
        int r = rb + i * 8;
        as[r * LDW + sc] = (_Float16)(va[i] * (fmA[i] * kmask));
        bs[r * LDW + sc] = (_Float16)(vb[i] * (fmB[i] * kmask));
      }
    };

    const int nt = (K + BK - 1) / BK;
    float va[16], vb[16];
    loads(0, va, vb);
    stores(As[0], Bs[0], 0, va, vb);
    __syncthreads();
    for (int t = 0; t < nt; ++t) {
      const int cur = t & 1;
      const bool more = (t + 1) < nt;
      if (more) loads((t + 1) * BK, va, vb);
      do_tile(As[cur], Bs[cur]);
      if (more) {
        stores(As[cur ^ 1], Bs[cur ^ 1], (t + 1) * BK, va, vb);
        __syncthreads();
      }
    }
  }

  // Epilogue: C VGPR r -> m = r + 8*(lane>=16), n = lane%16
  #pragma unroll
  for (int tm = 0; tm < 2; ++tm) {
    #pragma unroll
    for (int tn = 0; tn < 4; ++tn) {
      int n = n0 + wn * 64 + tn * 16 + l16;
      #pragma unroll
      for (int r = 0; r < 8; ++r) {
        int m = m0 + wm * 32 + tm * 16 + lh * 8 + r;
        if (m < M && n < N) {
          float v = acc[tm][tn][r] * alpha;
          if (bias) v += bias[n];
          long long off = (flags & GF_TRANS) ? ((long long)n * ldc + m)
                                             : ((long long)m * ldc + n);
          if (flags & GF_ACC) v += C[off];
          if (flags & GF_RELU) v = fmaxf(v, 0.0f);
          C[off] = v;
        }
      }
    }
  }
}

// ---------------------------------------------------------------------------
// Row softmax over [rows, width]; one 256-thread block (8 wave32) per row.
// ---------------------------------------------------------------------------
__global__ __launch_bounds__(TB)
void softmax_rows(float* __restrict__ S, int width)
{
  __shared__ float red[8];
  long long base = (long long)blockIdx.x * width;
  int tid = threadIdx.x, lane = tid & 31, wv = tid >> 5;
  float m = -3.4e38f;
  for (int i = tid; i < width; i += TB) m = fmaxf(m, S[base + i]);
  for (int off = 16; off; off >>= 1) m = fmaxf(m, __shfl_xor(m, off, 32));
  if (lane == 0) red[wv] = m;
  __syncthreads();
  m = red[0];
  #pragma unroll
  for (int k = 1; k < 8; ++k) m = fmaxf(m, red[k]);
  __syncthreads();
  float s = 0.0f;
  for (int i = tid; i < width; i += TB) {
    float e = expf(S[base + i] - m);
    S[base + i] = e;
    s += e;
  }
  for (int off = 16; off; off >>= 1) s += __shfl_xor(s, off, 32);
  if (lane == 0) red[wv] = s;
  __syncthreads();
  s = 0.0f;
  #pragma unroll
  for (int k = 0; k < 8; ++k) s += red[k];
  float inv = 1.0f / s;
  for (int i = tid; i < width; i += TB) S[base + i] *= inv;
}

// ---------------------------------------------------------------------------
// x = LayerNorm(x + r) * g + b, width fixed to 1024 (D). One block per row.
// ---------------------------------------------------------------------------
__global__ __launch_bounds__(TB)
void add_layernorm(float* __restrict__ x, const float* __restrict__ r,
                   const float* __restrict__ g, const float* __restrict__ b)
{
  __shared__ float rs[8], rs2[8];
  long long base = (long long)blockIdx.x * 1024;
  int tid = threadIdx.x, lane = tid & 31, wv = tid >> 5;
  float v[4], s = 0.0f, s2 = 0.0f;
  #pragma unroll
  for (int j = 0; j < 4; ++j) {
    int i = tid + j * TB;
    float t = x[base + i] + r[base + i];
    v[j] = t; s += t; s2 += t * t;
  }
  for (int off = 16; off; off >>= 1) { s += __shfl_xor(s, off, 32); s2 += __shfl_xor(s2, off, 32); }
  if (lane == 0) { rs[wv] = s; rs2[wv] = s2; }
  __syncthreads();
  s = 0.0f; s2 = 0.0f;
  #pragma unroll
  for (int k = 0; k < 8; ++k) { s += rs[k]; s2 += rs2[k]; }
  float mean = s * (1.0f / 1024.0f);
  float var  = s2 * (1.0f / 1024.0f) - mean * mean;
  float rstd = rsqrtf(var + 1e-5f);
  #pragma unroll
  for (int j = 0; j < 4; ++j) {
    int i = tid + j * TB;
    x[base + i] = (v[j] - mean) * rstd * g[i] + b[i];
  }
}

// ---------------------------------------------------------------------------
// Conv1d(k=2, pad=1) im2col: Xc[t, 2c+j] = x[t-1+j, c] (zeros out of range).
// ---------------------------------------------------------------------------
__global__ __launch_bounds__(TB)
void conv_pack(const float* __restrict__ x, float* __restrict__ Xc, int nin, int enc)
{
  long long idx = (long long)blockIdx.x * TB + threadIdx.x;
  long long tot = (long long)(nin + 1) * enc;
  if (idx >= tot) return;
  int t = (int)(idx / enc);
  int c = (int)(idx % enc);
  float a = (t >= 1)  ? x[(long long)(t - 1) * enc + c] : 0.0f;
  float b = (t < nin) ? x[(long long)t * enc + c]       : 0.0f;
  long long o = (long long)t * (2 * enc) + 2 * c;
  Xc[o]     = a;
  Xc[o + 1] = b;
}

// ---------------------------------------------------------------------------
// Sin/cos positional tables: absolute and rel-resampled (round half-to-even).
// ---------------------------------------------------------------------------
__global__ __launch_bounds__(TB)
void build_pe(float* __restrict__ PEa, float* __restrict__ PEr, int n, int d, int rel)
{
  long long idx = (long long)blockIdx.x * TB + threadIdx.x;
  long long tot = (long long)n * d;
  if (idx >= tot) return;
  int t = (int)(idx / d);
  int c = (int)(idx % d);
  float div = expf(-logf(10000.0f) * (float)(c & ~1) / (float)d);
  float pa = (float)t * div;
  PEa[idx] = (c & 1) ? cosf(pa) : sinf(pa);
  int denom = (n - 1 > 1) ? (n - 1) : 1;
  float tr = rintf((float)t * (float)(rel - 1) / (float)denom);
  float pr = tr * div;
  PEr[idx] = (c & 1) ? cosf(pr) : sinf(pr);
}

// ---------------------------------------------------------------------------
// xcat = [x | cummax-before | cummax-after] per column (pool_score features).
// ---------------------------------------------------------------------------
__global__ __launch_bounds__(TB)
void build_xcat(const float* __restrict__ x, float* __restrict__ xcat, int n, int d)
{
  int c = blockIdx.x * TB + threadIdx.x;
  if (c >= d) return;
  float run = 0.0f;
  for (int t = 0; t < n; ++t) {
    float xv = x[(long long)t * d + c];
    long long o = (long long)t * (3 * d);
    xcat[o + c]     = xv;
    xcat[o + d + c] = (t == 0) ? 0.0f : run;
    run = (t == 0) ? xv : fmaxf(run, xv);
  }
  run = 0.0f;
  for (int t = n - 1; t >= 0; --t) {
    float xv = x[(long long)t * d + c];
    long long o = (long long)t * (3 * d);
    xcat[o + 2 * d + c] = (t == n - 1) ? 0.0f : run;
    run = (t == n - 1) ? xv : fmaxf(run, xv);
  }
}

// ------------------------------ host side ----------------------------------
static inline void gemm(hipStream_t st,
                        const float* A, long long lda, long long sA,
                        const float* B, long long ldb, long long sB,
                        float* C, long long ldc, long long sC,
                        const float* bias, int M, int N, int K,
                        float alpha, int flags, int batch)
{
  dim3 grid((unsigned)((N + BN - 1) / BN), (unsigned)((M + BM - 1) / BM), (unsigned)batch);
  gemm_f16_wmma<<<grid, TB, 0, st>>>(A, lda, sA, B, ldb, sB, C, ldc, sC,
                                     bias, M, N, K, alpha, flags);
}

struct WS {
  float *Xc, *PEa, *PEr, *xc, *xm, *Q, *Kb, *Vt, *S, *AO, *T1, *T2, *XQ, *XK;
};

// mha(xq[nq,1024], xkv[nk,1024]) -> out[nq,1024]; params at din[pb..pb+7]
static void run_mha(hipStream_t st, const float* xq, int nq, const float* xkv, int nk,
                    void* const* din, int pb, const WS& w, float* out)
{
  const float* Wq = (const float*)din[pb + 0]; const float* bq = (const float*)din[pb + 1];
  const float* Wk = (const float*)din[pb + 2]; const float* bk = (const float*)din[pb + 3];
  const float* Wv = (const float*)din[pb + 4]; const float* bv = (const float*)din[pb + 5];
  const float* Wo = (const float*)din[pb + 6]; const float* bo = (const float*)din[pb + 7];

  gemm(st, xq,  1024, 0, Wq, 1024, 0, w.Q,  1024, 0, bq, nq, 1024, 1024, 1.0f, 0, 1);
  gemm(st, xkv, 1024, 0, Wk, 1024, 0, w.Kb, 1024, 0, bk, nk, 1024, 1024, 1.0f, 0, 1);
  // V projection stored transposed: Vt[1024, nk] so P@V stays K-contiguous.
  gemm(st, xkv, 1024, 0, Wv, 1024, 0, w.Vt, nk, 0, bv, nk, 1024, 1024, 1.0f, GF_TRANS, 1);

  const float iscale = 0.17677669529663687f;  // 1/sqrt(dh=32)
  for (int h0 = 0; h0 < 32; h0 += 8) {        // 8 heads per group (bounds S workspace)
    gemm(st, w.Q + h0 * 32, 1024, 32, w.Kb + h0 * 32, 1024, 32,
         w.S, nk, (long long)nq * nk, nullptr, nq, nk, 32, iscale, 0, 8);
    softmax_rows<<<(unsigned)(8 * nq), TB, 0, st>>>(w.S, nk);
    gemm(st, w.S, nk, (long long)nq * nk,
         w.Vt + (long long)h0 * 32 * nk, nk, (long long)32 * nk,
         w.AO + h0 * 32, 1024, 32, nullptr, nq, 32, nk, 1.0f, 0, 8);
  }
  gemm(st, w.AO, 1024, 0, Wo, 1024, 0, out, 1024, 0, bo, nq, 1024, 1024, 1.0f, 0, 1);
}

// post-norm TransformerDecoderLayer; params flattened at din[base..base+25]
static void run_dec_layer(hipStream_t st, float* x, int n, const float* mem, int nm,
                          void* const* din, int base, const WS& w)
{
  run_mha(st, x, n, x, n, din, base + 0, w, w.T2);
  add_layernorm<<<(unsigned)n, TB, 0, st>>>(x, w.T2,
      (const float*)din[base + 20], (const float*)din[base + 21]);
  run_mha(st, x, n, mem, nm, din, base + 8, w, w.T2);
  add_layernorm<<<(unsigned)n, TB, 0, st>>>(x, w.T2,
      (const float*)din[base + 22], (const float*)din[base + 23]);
  gemm(st, x, 1024, 0, (const float*)din[base + 16], 1024, 0, w.T1, 2048, 0,
       (const float*)din[base + 17], n, 2048, 1024, 1.0f, GF_RELU, 1);
  gemm(st, w.T1, 2048, 0, (const float*)din[base + 18], 2048, 0, w.T2, 1024, 0,
       (const float*)din[base + 19], n, 1024, 2048, 1.0f, 0, 1);
  add_layernorm<<<(unsigned)n, TB, 0, st>>>(x, w.T2,
      (const float*)din[base + 24], (const float*)din[base + 25]);
}

extern "C" void kernel_launch(void* const* d_in, const int* in_sizes, int n_in,
                              void* d_out, int out_size, void* d_ws, size_t ws_size,
                              hipStream_t stream)
{
  (void)in_sizes; (void)n_in; (void)out_size; (void)ws_size;
  const int NC = 1025, NM = 512;

  // ---- workspace carve (floats), ~103 MB total ----
  WS w;
  float* p = (float*)d_ws;
  auto take = [&](long long nelem) { float* q = p; p += nelem; return q; };
  w.Xc  = take((long long)NC * 3072);   // conv im2col; reused for xcat
  w.PEa = take((long long)NC * 1024);
  w.PEr = take((long long)NC * 1024);
  w.xc  = take((long long)NC * 1024);
  w.xm  = take((long long)NM * 1024);
  w.Q   = take((long long)NC * 1024);
  w.Kb  = take((long long)NC * 1024);
  w.Vt  = take((long long)NC * 1024);   // [1024, nk] transposed V
  w.S   = take(8LL * NC * NC);          // 8-head score group
  w.AO  = take((long long)NC * 1024);
  w.T1  = take((long long)NC * 2048);
  w.T2  = take((long long)NC * 1024);   // also holds [512,2048] pool intermediate
  w.XQ  = take((long long)NM * 2048);
  w.XK  = take((long long)NC * 2048);

  auto F = [&](int i) { return (const float*)d_in[i]; };
  // DFS-flattened input indices:
  // 0 x_code_pooled, 1 x_md_pooled, 2 conv_w, 3 conv_b, 4/5 md2dec,
  // 6/7 abs_pe, 8/9 rel_pe, 10+26*l code_dec[l], 114+26*l md_dec[l],
  // 218..225 pool(q1,q2,k1,k2), 226..229 between, 230..233 after_md

  // Conv1d front-end as a single K=3072 GEMM
  {
    long long tot = (long long)NC * 1536;
    conv_pack<<<(unsigned)((tot + TB - 1) / TB), TB, 0, stream>>>(F(0), w.Xc, 1024, 1536);
  }
  gemm(stream, w.Xc, 3072, 0, F(2), 3072, 0, w.xc, 1024, 0, F(3), NC, 1024, 3072, 1.0f, 0, 1);

  // Combined positional encoding: x += PEabs@Wa^T+ba  += PErel@Wr^T+br
  {
    long long tot = (long long)NC * 1024;
    build_pe<<<(unsigned)((tot + TB - 1) / TB), TB, 0, stream>>>(w.PEa, w.PEr, NC, 1024, 1024);
  }
  gemm(stream, w.PEa, 1024, 0, F(6), 1024, 0, w.xc, 1024, 0, F(7), NC, 1024, 1024, 1.0f, GF_ACC, 1);
  gemm(stream, w.PEr, 1024, 0, F(8), 1024, 0, w.xc, 1024, 0, F(9), NC, 1024, 1024, 1.0f, GF_ACC, 1);

  // md2dec projection
  gemm(stream, F(1), 1536, 0, F(4), 1536, 0, w.xm, 1024, 0, F(5), NM, 1024, 1536, 1.0f, 0, 1);

  // Interleaved decoder stacks
  for (int l = 0; l < 4; ++l) {
    run_dec_layer(stream, w.xc, NC, w.xm, NM, d_in, 10 + l * 26, w);
    run_dec_layer(stream, w.xm, NM, w.xc, NC, d_in, 114 + l * 26, w);
  }

  float* out = (float*)d_out;
  const float hs = 0.022097086912079608f;  // 1/sqrt(HID=2048)

  // pool_score -> out[0 : 512*1025]
  build_xcat<<<4, TB, 0, stream>>>(w.xc, w.Xc, NC, 1024);
  gemm(stream, w.Xc, 3072, 0, F(222), 3072, 0, w.T1, 2048, 0, F(223), NC, 2048, 3072, 1.0f, GF_RELU, 1);
  gemm(stream, w.T1, 2048, 0, F(224), 2048, 0, w.XK, 2048, 0, F(225), NC, 2048, 2048, 1.0f, 0, 1);
  gemm(stream, w.xm, 1024, 0, F(218), 1024, 0, w.T2, 2048, 0, F(219), NM, 2048, 1024, 1.0f, GF_RELU, 1);
  gemm(stream, w.T2, 2048, 0, F(220), 2048, 0, w.XQ, 2048, 0, F(221), NM, 2048, 2048, 1.0f, 0, 1);
  gemm(stream, w.XQ, 2048, 0, w.XK, 2048, 0, out, NC, 0, nullptr, NM, NC, 2048, hs, 0, 1);

  // ca_score "between" -> out[512*1025 : 2*512*1025]
  gemm(stream, w.xm, 1024, 0, F(226), 1024, 0, w.XQ, 2048, 0, F(227), NM, 2048, 1024, 1.0f, 0, 1);
  gemm(stream, w.xc, 1024, 0, F(228), 1024, 0, w.XK, 2048, 0, F(229), NC, 2048, 1024, 1.0f, 0, 1);
  gemm(stream, w.XQ, 2048, 0, w.XK, 2048, 0, out + (long long)NM * NC, NC, 0,
       nullptr, NM, NC, 2048, hs, 0, 1);

  // ca_score "after_md" -> out[2*512*1025 : +512*512]
  gemm(stream, w.xm, 1024, 0, F(230), 1024, 0, w.XQ, 2048, 0, F(231), NM, 2048, 1024, 1.0f, 0, 1);
  gemm(stream, w.xm, 1024, 0, F(232), 1024, 0, w.XK, 2048, 0, F(233), NM, 2048, 1024, 1.0f, 0, 1);
  gemm(stream, w.XQ, 2048, 0, w.XK, 2048, 0, out + 2LL * NM * NC, NM, 0,
       nullptr, NM, NM, 2048, hs, 0, 1);
}
